// AttnBlock_61435212202711
// MI455X (gfx1250) — matrix operand order repaired and checked by reading, not compiled, not run on hardware
//
#include <hip/hip_runtime.h>
#include <math.h>

// ---------------------------------------------------------------------------
// Types for CDNA5 WMMA (wave32). v_wmma_f32_16x16x32_bf16: D=A(16x32)xB(32x16)+C
// ---------------------------------------------------------------------------
typedef __attribute__((ext_vector_type(16))) __bf16 v16bf;
typedef __attribute__((ext_vector_type(8)))  __bf16 v8bf;
typedef __attribute__((ext_vector_type(8)))  float  v8f;

__device__ __forceinline__ unsigned short f32_to_bf16(float f) {
  union { float f; unsigned u; } v; v.f = f;
  unsigned r = v.u + 0x7FFFu + ((v.u >> 16) & 1u);   // round-to-nearest-even
  return (unsigned short)(r >> 16);
}

// A-fragment: lane holds row m=lane&15; halves 0..7 = K+0..7 (lanes<16) or K+8..15,
// halves 8..15 = those +16.  Two aligned b128 loads.
__device__ __forceinline__ v16bf load_frag_a(const unsigned short* p) {
  union { v16bf v; v8bf h[2]; } u;
  u.h[0] = *(const v8bf*)(p);
  u.h[1] = *(const v8bf*)(p + 16);
  return u.v;
}
// B-fragment: lane holds col n=lane&15; halves 0..7 = K+0..7 (lanes<16) or K+8..15,
// halves 8..15 = those +16.
__device__ __forceinline__ v16bf load_frag_b(const unsigned short* p) {
  union { v16bf v; v8bf h[2]; } u;
  u.h[0] = *(const v8bf*)(p);
  u.h[1] = *(const v8bf*)(p + 16);
  return u.v;
}

// ---------------------------------------------------------------------------
// Generic GEMM:  Out[m,n] = scale * (sum_k A[m,k]*Bt[n,k] + bias[m]) (+res)
// Block = 256 thr = 8 waves (2 in M x 4 in N). Wave tile 64x64 (4x4 subtiles of
// 16x16), K step 32 -> 16 v_wmma per step, operands streamed from L2/HBM.
// MODE: 0 = bf16 transposed store OutT[n][m] (packed b128)
//       1 = bf16 normal store Out[m][n]
//       2 = f32  normal store
//       3 = f32  normal store + residual add
// ---------------------------------------------------------------------------
template<int MODE, bool HAS_BIAS>
__global__ __launch_bounds__(256)
void gemm_abt(const unsigned short* __restrict__ A,  int lda, long long sA,
              const unsigned short* __restrict__ Bt, int ldb, long long sB,
              void* __restrict__ OutV,               int ldo, long long sO,
              const float* __restrict__ bias,
              const float* __restrict__ res, long long sR,
              float scale, int K)
{
  const int tid  = threadIdx.x;
  const int lane = tid & 31;
  const int w    = tid >> 5;
  const int m0   = blockIdx.y * 128 + (w & 1) * 64;
  const int n0   = blockIdx.x * 256 + (w >> 1) * 64;

  const unsigned short* Ab  = A  + (size_t)blockIdx.z * (size_t)sA;
  const unsigned short* Btb = Bt + (size_t)blockIdx.z * (size_t)sB;

  const int arow  = lane & 15;
  const int akoff = (lane < 16) ? 0 : 8;
  const int bcol  = lane & 15;
  const int bkoff = (lane < 16) ? 0 : 8;

  v8f acc[4][4] = {};

  for (int kb = 0; kb < K; kb += 32) {
    v16bf af[4], bfv[4];
#pragma unroll
    for (int mi = 0; mi < 4; ++mi) {
      const unsigned short* p = Ab + (size_t)(m0 + mi * 16 + arow) * lda + kb + akoff;
      af[mi] = load_frag_a(p);
      __builtin_prefetch(p + 32, 0, 3);        // global_prefetch_b8: next K slab
    }
#pragma unroll
    for (int ni = 0; ni < 4; ++ni) {
      const unsigned short* p = Btb + (size_t)(n0 + ni * 16 + bcol) * ldb + kb + bkoff;
      bfv[ni] = load_frag_b(p);
      __builtin_prefetch(p + 32, 0, 3);
    }
#pragma unroll
    for (int mi = 0; mi < 4; ++mi)
#pragma unroll
      for (int ni = 0; ni < 4; ++ni)
        acc[mi][ni] = __builtin_amdgcn_wmma_f32_16x16x32_bf16(
            false, af[mi], false, bfv[ni], (short)0, acc[mi][ni], false, false);
  }

  // C/D layout: lane col n = lane&15, VGPR r -> row m = (lane<16?0:8)+r
  const int ccol  = lane & 15;
  const int rbase = (lane < 16) ? 0 : 8;
#pragma unroll
  for (int mi = 0; mi < 4; ++mi) {
    const int mrow = m0 + mi * 16 + rbase;
#pragma unroll
    for (int ni = 0; ni < 4; ++ni) {
      const int n = n0 + ni * 16 + ccol;
      if constexpr (MODE == 0) {
        unsigned short* O = (unsigned short*)OutV + (size_t)blockIdx.z * (size_t)sO;
        union { v8bf v; unsigned short s[8]; } pk;
#pragma unroll
        for (int r = 0; r < 8; ++r) {
          float vv = acc[mi][ni][r];
          if constexpr (HAS_BIAS) vv += bias[mrow + r];
          pk.s[r] = f32_to_bf16(vv * scale);
        }
        *(v8bf*)(O + (size_t)n * ldo + mrow) = pk.v;   // packed b128 store
      } else if constexpr (MODE == 1) {
        unsigned short* O = (unsigned short*)OutV + (size_t)blockIdx.z * (size_t)sO;
#pragma unroll
        for (int r = 0; r < 8; ++r) {
          float vv = acc[mi][ni][r];
          if constexpr (HAS_BIAS) vv += bias[mrow + r];
          O[(size_t)(mrow + r) * ldo + n] = f32_to_bf16(vv * scale);
        }
      } else {
        float* O = (float*)OutV + (size_t)blockIdx.z * (size_t)sO;
#pragma unroll
        for (int r = 0; r < 8; ++r) {
          float vv = acc[mi][ni][r];
          if constexpr (HAS_BIAS) vv += bias[mrow + r];
          vv *= scale;
          if constexpr (MODE == 3)
            vv += res[(size_t)blockIdx.z * (size_t)sR + (size_t)(mrow + r) * ldo + n];
          O[(size_t)(mrow + r) * ldo + n] = vv;
        }
      }
    }
  }
}

// ---------------------------------------------------------------------------
// GroupNorm: one block per (batch, group); 16 channels x 4096 positions.
// Writes hnT [B][N][C] in bf16 (transposed so projections get contiguous K).
// ---------------------------------------------------------------------------
__global__ __launch_bounds__(256)
void groupnorm_kernel(const float* __restrict__ x, const float* __restrict__ gamma,
                      const float* __restrict__ beta, unsigned short* __restrict__ hnT)
{
  const int b  = blockIdx.x >> 5;
  const int g  = blockIdx.x & 31;
  const int c0 = g * 16;
  const int tid = threadIdx.x;
  const float* xb = x + ((size_t)b * 512 + c0) * 4096;

  float s = 0.f, ss = 0.f;
  for (int idx = tid; idx < 16 * 4096; idx += 256) {   // contiguous, coalesced
    float v = xb[idx];
    s += v; ss += v * v;
  }
  __shared__ float r1[256], r2[256];
  r1[tid] = s; r2[tid] = ss;
  __syncthreads();
  for (int off = 128; off > 0; off >>= 1) {
    if (tid < off) { r1[tid] += r1[tid + off]; r2[tid] += r2[tid + off]; }
    __syncthreads();
  }
  const float inv  = 1.0f / 65536.0f;
  const float mean = r1[0] * inv;
  const float var  = r2[0] * inv - mean * mean;
  const float rstd = rsqrtf(var + 1e-6f);

  unsigned short* ho = hnT + (size_t)b * 4096 * 512 + c0;
  for (int idx = tid; idx < 16 * 4096; idx += 256) {
    const int c = idx & 15;
    const int n = idx >> 4;
    const float v = xb[(size_t)c * 4096 + n];          // L2-hot second pass
    const float y = (v - mean) * rstd * gamma[c0 + c] + beta[c0 + c];
    ho[(size_t)n * 512 + c] = f32_to_bf16(y);
  }
}

// ---------------------------------------------------------------------------
// Row softmax over S [B][4096][4096] fp32; rewrites each row in place as bf16
// probabilities with row stride 8192 halfwords (reusing the fp32 buffer).
// ---------------------------------------------------------------------------
__global__ __launch_bounds__(256)
void softmax_kernel(float* __restrict__ S)
{
  const size_t row = (size_t)blockIdx.x + (size_t)blockIdx.y * 4096;
  float* Sr = S + row * 4096;
  unsigned short* Pr = (unsigned short*)Sr;
  const int tid = threadIdx.x;

  float f[16];
  float mx = -3.4e38f;
#pragma unroll
  for (int t = 0; t < 16; ++t) { f[t] = Sr[tid + t * 256]; mx = fmaxf(mx, f[t]); }

  __shared__ float red[256];
  red[tid] = mx; __syncthreads();
  for (int off = 128; off > 0; off >>= 1) {
    if (tid < off) red[tid] = fmaxf(red[tid], red[tid + off]);
    __syncthreads();
  }
  mx = red[0];
  __syncthreads();

  float s = 0.f;
#pragma unroll
  for (int t = 0; t < 16; ++t) { f[t] = __expf(f[t] - mx); s += f[t]; }
  red[tid] = s; __syncthreads();
  for (int off = 128; off > 0; off >>= 1) {
    if (tid < off) red[tid] += red[tid + off];
    __syncthreads();
  }
  const float invs = 1.0f / red[0];
  // All reads of this row finished (barriers above) -> safe in-place rewrite.
#pragma unroll
  for (int t = 0; t < 16; ++t) Pr[tid + t * 256] = f32_to_bf16(f[t] * invs);
}

__global__ __launch_bounds__(256)
void cvt_f32_bf16(const float* __restrict__ src, unsigned short* __restrict__ dst, int n)
{
  int i = blockIdx.x * blockDim.x + threadIdx.x;
  if (i < n) dst[i] = f32_to_bf16(src[i]);
}

// ---------------------------------------------------------------------------
// Workspace layout (bytes). Total ~170 MiB.
// ---------------------------------------------------------------------------
static constexpr long long NC       = 4096LL * 512;          // elems per batch slab
static constexpr size_t SLAB        = (size_t)NC * 2 * 2;    // B=2, bf16
static constexpr size_t OFF_HNT     = 0;
static constexpr size_t OFF_QT      = OFF_HNT + SLAB;
static constexpr size_t OFF_KT      = OFF_QT  + SLAB;
static constexpr size_t OFF_V       = OFF_KT  + SLAB;
static constexpr size_t OFF_AOT     = OFF_V   + SLAB;
static constexpr size_t OFF_WQ      = OFF_AOT + SLAB;
static constexpr size_t OFF_WK      = OFF_WQ + 512 * 512 * 2;
static constexpr size_t OFF_WV      = OFF_WK + 512 * 512 * 2;
static constexpr size_t OFF_WO      = OFF_WV + 512 * 512 * 2;
static constexpr size_t OFF_S       = OFF_WO + 512 * 512 * 2; // fp32 [2][4096][4096]

extern "C" void kernel_launch(void* const* d_in, const int* in_sizes, int n_in,
                              void* d_out, int out_size, void* d_ws, size_t ws_size,
                              hipStream_t stream) {
  (void)in_sizes; (void)n_in; (void)out_size; (void)ws_size;
  const float* x     = (const float*)d_in[0];
  const float* gamma = (const float*)d_in[1];
  const float* beta  = (const float*)d_in[2];
  const float* Wq = (const float*)d_in[3]; const float* bq = (const float*)d_in[4];
  const float* Wk = (const float*)d_in[5]; const float* bk = (const float*)d_in[6];
  const float* Wv = (const float*)d_in[7]; const float* bv = (const float*)d_in[8];
  const float* Wo = (const float*)d_in[9]; const float* bo = (const float*)d_in[10];
  float* out = (float*)d_out;

  char* ws = (char*)d_ws;
  unsigned short* hnT = (unsigned short*)(ws + OFF_HNT);   // [B][N][C] bf16
  unsigned short* qT  = (unsigned short*)(ws + OFF_QT);    // [B][N][C] bf16 (pre-scaled)
  unsigned short* kT  = (unsigned short*)(ws + OFF_KT);    // [B][N][C] bf16
  unsigned short* v   = (unsigned short*)(ws + OFF_V);     // [B][C][N] bf16
  unsigned short* aoT = (unsigned short*)(ws + OFF_AOT);   // [B][N][C] bf16
  unsigned short* wqb = (unsigned short*)(ws + OFF_WQ);
  unsigned short* wkb = (unsigned short*)(ws + OFF_WK);
  unsigned short* wvb = (unsigned short*)(ws + OFF_WV);
  unsigned short* wob = (unsigned short*)(ws + OFF_WO);
  float*          S   = (float*)(ws + OFF_S);              // [B][4096][4096] fp32

  const int NW = 512 * 512;
  cvt_f32_bf16<<<(NW + 255) / 256, 256, 0, stream>>>(Wq, wqb, NW);
  cvt_f32_bf16<<<(NW + 255) / 256, 256, 0, stream>>>(Wk, wkb, NW);
  cvt_f32_bf16<<<(NW + 255) / 256, 256, 0, stream>>>(Wv, wvb, NW);
  cvt_f32_bf16<<<(NW + 255) / 256, 256, 0, stream>>>(Wo, wob, NW);

  groupnorm_kernel<<<64, 256, 0, stream>>>(x, gamma, beta, hnT);

  const float qscale = 1.0f / sqrtf(512.0f);
  dim3 gp(4096 / 256, 512 / 128, 2);     // projections / PV: M=512, N=4096
  // q = (Wq·h + bq) * scale  -> qT [N][C]
  gemm_abt<0, true ><<<gp, 256, 0, stream>>>(wqb, 512, 0, hnT, 512, NC,
                                             qT, 512, NC, bq, nullptr, 0, qscale, 512);
  // k -> kT [N][C]
  gemm_abt<0, true ><<<gp, 256, 0, stream>>>(wkb, 512, 0, hnT, 512, NC,
                                             kT, 512, NC, bk, nullptr, 0, 1.0f, 512);
  // v -> [C][N]
  gemm_abt<1, true ><<<gp, 256, 0, stream>>>(wvb, 512, 0, hnT, 512, NC,
                                             v, 4096, NC, bv, nullptr, 0, 1.0f, 512);

  // scores S[i][j] = qT[i,:]·kT[j,:]  (scale already in q), fp32
  dim3 gs(4096 / 256, 4096 / 128, 2);
  gemm_abt<2, false><<<gs, 256, 0, stream>>>(qT, 512, NC, kT, 512, NC,
                                             S, 4096, 4096LL * 4096, nullptr,
                                             nullptr, 0, 1.0f, 512);

  dim3 gsm(4096, 2);
  softmax_kernel<<<gsm, 256, 0, stream>>>(S);   // -> P bf16 in place, row stride 8192

  // aoT[i][c] = sum_j P[i][j] * v[c][j]   (A=v [C][N], Bt=P [N][8192], K=4096)
  gemm_abt<0, false><<<gp, 256, 0, stream>>>(v, 4096, NC, (unsigned short*)S, 8192,
                                             4096LL * 8192, aoT, 512, NC, nullptr,
                                             nullptr, 0, 1.0f, 4096);

  // out = x + Wo·ao + bo   (fp32, fused residual)
  gemm_abt<3, true ><<<gp, 256, 0, stream>>>(wob, 512, 0, aoT, 512, NC,
                                             out, 4096, NC, bo, x, NC, 1.0f, 512);
}
